// NeRF_35175782154243
// MI455X (gfx1250) — matrix operand order repaired and checked by reading, not compiled, not run on hardware
//
#include <hip/hip_runtime.h>

typedef _Float16 v16h  __attribute__((ext_vector_type(16)));
typedef float    v8f   __attribute__((ext_vector_type(8)));
typedef float    f32x4 __attribute__((ext_vector_type(4)));   // POD 16-byte vector

// LDS (address_space 3) typed pointers: 32-bit arithmetic, no null checks.
typedef __attribute__((address_space(3))) _Float16 ldsh;
typedef __attribute__((address_space(3))) f32x4    ldsf4;

union F16x16 { v16h h; f32x4 f[2]; };

// ---------------- LDS layout (units of _Float16) ----------------
// 132 KB per workgroup -> two workgroups co-resident per 320 KB WGP
// -> 4 wave32s per SIMD to hide LDS/WMMA latency.
constexpr int ROWS         = 64;    // rows (points) per workgroup
constexpr int EPTS_STRIDE  = 112;   // 96 cols + 16 pad
constexpr int EVIEW_STRIDE = 80;    // 64 cols + 16 pad
constexpr int H_STRIDE     = 272;   // 256 cols + 16 pad
constexpr int WB_STRIDE    = 40;    // 32 K + 8 pad (column-major weight slab)
constexpr int EPTS_OFF   = 0;
constexpr int EVIEW_OFF  = EPTS_OFF  + ROWS * EPTS_STRIDE;   // 7168
constexpr int HA_OFF     = EVIEW_OFF + ROWS * EVIEW_STRIDE;  // 12288
constexpr int HB_OFF     = HA_OFF    + ROWS * H_STRIDE;      // 29696
constexpr int WB_OFF     = HB_OFF    + ROWS * H_STRIDE;      // 47104
constexpr int WB_SLAB    = 256 * WB_STRIDE;                  // 10240
constexpr int LDS_ELEMS  = WB_OFF + 2 * WB_SLAB;             // 67584 f16 = 132 KB

// ---------------- f16 weight arena in d_ws (element offsets) ----------------
constexpr int WSO_W0   = 0;                       // 256 x  96
constexpr int WSO_W1   = WSO_W0 + 256 * 96;       // 256 x 256
constexpr int WSO_W2   = WSO_W1 + 256 * 256;
constexpr int WSO_W3   = WSO_W2 + 256 * 256;
constexpr int WSO_W4   = WSO_W3 + 256 * 256;
constexpr int WSO_W5   = WSO_W4 + 256 * 256;      // 256 x 352
constexpr int WSO_W6   = WSO_W5 + 256 * 352;
constexpr int WSO_W7   = WSO_W6 + 256 * 256;
constexpr int WSO_FEAT = WSO_W7 + 256 * 256;      // 256 x 256
constexpr int WSO_VIEW = WSO_FEAT + 256 * 256;    // 128 x 320
constexpr int WSO_END  = WSO_VIEW + 128 * 320;    // 614400 elems = 1.2 MB

struct NerfArgs {
  const float* input_pts;
  const float* input_views;
  const float* input_pls;
  const _Float16* wf[8];          // f16 weights in ws
  const float* b[8];
  const _Float16* views_wf;  const float* views_b;
  const _Float16* feature_wf;const float* feature_b;
  const float* alpha_w;      const float* alpha_b;
  const float* rgb_w;        const float* rgb_b;
};

// -------- fp32 -> f16 weight repack (row-major [Ksrc][ncols] -> col-major
// [ncols][Kpad] with a zero gap inserted at [insert_at, insert_at+gap)) -------
__global__ void conv_weights(const float* __restrict__ src, _Float16* __restrict__ dst,
                             int ncols, int Kpad, int Ksrc, int insert_at, int gap) {
  const int k = blockIdx.x * blockDim.x + threadIdx.x;
  const int n = blockIdx.y;
  if (k >= Kpad) return;
  const int srck = (k < insert_at) ? k : (k - gap);
  const bool ok = (k < insert_at || k >= insert_at + gap) && (srck < Ksrc);
  const float v = ok ? src[srck * ncols + n] : 0.f;
  dst[n * Kpad + k] = (_Float16)v;
}

// A fragment for V_WMMA_F32_16X16X32_F16: lanes 0-15 row M=lane,
// K = {kb..kb+7, kb+16..kb+23}; lanes 16-31 the +8-shifted chunks.
__device__ inline v16h load_a(const ldsh* p, int sel) {
  F16x16 u;
  u.f[0] = *(const ldsf4*)(p + sel * 8);
  u.f[1] = *(const ldsf4*)(p + 16 + sel * 8);
  return u.h;
}

// B fragment: lane holds column N (=lc within tile), K = {sel*8..+7, 16+sel*8..+7}.
__device__ inline v16h load_b(const ldsh* p) {
  F16x16 u;
  u.f[0] = *(const ldsf4*)(p);
  u.f[1] = *(const ldsf4*)(p + 16);
  return u.h;
}

// DMA one K=32 weight slab (ncols*64 B) from ws (f16, col-major, stride Kpad)
// into the LDS slab at byte offset ldsByte (CDNA5 async-to-LDS, ASYNCcnt).
template<int NTOT>
__device__ inline void stage_async(const _Float16* __restrict__ Wf, int Kpad, int ks,
                                   unsigned ldsByte) {
  const int tid = threadIdx.x;
#pragma unroll
  for (int j = 0; j < NTOT / 4; ++j) {           // NTOT*64 chunks / 256 threads
    const int e = tid + j * 256;
    const int n = e >> 2;
    const int q = e & 3;
    const unsigned goff = (unsigned)(n * (Kpad * 2) + ks * 64 + q * 16);
    const unsigned loff = ldsByte + (unsigned)(n * (WB_STRIDE * 2) + q * 16);
    asm volatile("global_load_async_to_lds_b128 %0, %1, %2 offset:0"
                 :: "v"(loff), "v"(goff), "s"(Wf) : "memory");
  }
}

__device__ inline void wait_async0() {
  asm volatile("s_wait_asynccnt 0x0" ::: "memory");
}

// Fused GEMM layer: out[ROWS x NTOT*16] = act(A[ROWS x Kpad] * Wf + bias).
// Wave w owns rows (w&1)*32 (2 M-tiles) and N-tile quarter (w>>1): each B
// fragment feeds 2 WMMAs, each A fragment NTOT/4 WMMAs. acc = 64 VGPRs ->
// multi-wave residency per SIMD hides the ds_load latency.
// A fragments for the next k-step are prefetched before the slab-handoff barrier.
template<int NTOT, bool RELU>
__device__ void gemm_layer(const _Float16* __restrict__ Wf,
                           const float* __restrict__ bias, int Kpad,
                           const ldsh* a0, int lda0, int limit,
                           const ldsh* a1, int lda1,
                           ldsh* outH, int ldo,
                           ldsh* wbuf) {
  constexpr int NT = NTOT / 4;
  const int tid  = threadIdx.x;
  const int lane = tid & 31;
  const int wv   = tid >> 5;
  const int sel  = lane >> 4;
  const int lc   = lane & 15;
  const int rowBase = (wv & 1) * 32;
  const int ntOff   = (wv >> 1) * NT;
  const int nks  = Kpad >> 5;
  const unsigned wbufLds = (unsigned)(uintptr_t)wbuf;

  v8f acc[2][NT];
#pragma unroll
  for (int nt = 0; nt < NT; ++nt) {
    float bv = bias[(ntOff + nt) * 16 + lc];
    v8f t0 = {bv, bv, bv, bv, bv, bv, bv, bv};
    acc[0][nt] = t0;
    acc[1][nt] = t0;
  }

  // Per-m-tile A row pointers for both sources
  const ldsh* a0r0 = a0 + (rowBase + lc) * lda0;
  const ldsh* a0r1 = a0 + (rowBase + 16 + lc) * lda0;
  const ldsh* a1r0 = a1 + (rowBase + lc) * lda1;
  const ldsh* a1r1 = a1 + (rowBase + 16 + lc) * lda1;

  stage_async<NTOT>(Wf, Kpad, 0, wbufLds);

  // A fragments for ks=0 (activation buffers are stable for the whole layer)
  v16h A0 = load_a(a0r0, sel);
  v16h A1 = load_a(a0r1, sel);

  wait_async0();
  __syncthreads();

#pragma unroll 1
  for (int ks = 0; ks < nks; ++ks) {
    if (ks + 1 < nks)                               // DMA next slab while computing
      stage_async<NTOT>(Wf, Kpad, ks + 1, wbufLds + ((ks + 1) & 1) * (WB_SLAB * 2));

    const ldsh* wb = wbuf + (ks & 1) * WB_SLAB + (ntOff * 16 + lc) * WB_STRIDE + sel * 8;

    // Load the k-step's B fragments into distinct registers first so the
    // ds_load pairs can issue back-to-back and retire under the WMMAs.
    v16h B[NT];
#pragma unroll
    for (int nt = 0; nt < NT; ++nt)
      B[nt] = load_b(wb + nt * (16 * WB_STRIDE));

#pragma unroll
    for (int nt = 0; nt < NT; ++nt) {
      acc[0][nt] = __builtin_amdgcn_wmma_f32_16x16x32_f16(
          false, A0, false, B[nt], (short)0, acc[0][nt], false, false);
      acc[1][nt] = __builtin_amdgcn_wmma_f32_16x16x32_f16(
          false, A1, false, B[nt], (short)0, acc[1][nt], false, false);
    }

    // Prefetch A for ks+1 (crosses the barrier legally: A buffers are read-only here)
    if (ks + 1 < nks) {
      const int kb = (ks + 1) * 32;
      if (kb < limit) {                             // wave-uniform branch
        A0 = load_a(a0r0 + kb, sel);
        A1 = load_a(a0r1 + kb, sel);
      } else {
        const int kk = kb - limit;
        A0 = load_a(a1r0 + kk, sel);
        A1 = load_a(a1r1 + kk, sel);
      }
    }

    wait_async0();                                  // next slab landed in LDS
    __syncthreads();
  }

  // C/D layout: VGPR r, lane l -> M = r + 8*(l>>4), N = l&15
#pragma unroll
  for (int mt = 0; mt < 2; ++mt) {
#pragma unroll
    for (int nt = 0; nt < NT; ++nt) {
#pragma unroll
      for (int r = 0; r < 8; ++r) {
        float v = acc[mt][nt][r];
        if (RELU) v = v > 0.f ? v : 0.f;
        outH[(rowBase + mt * 16 + r + 8 * sel) * ldo + (ntOff + nt) * 16 + lc] =
            (_Float16)v;
      }
    }
  }
  __syncthreads();
}

// NeRF positional encoding straight into LDS (f16), zero-padded to 96 / 64 cols.
__device__ void encode_rows(const NerfArgs& A, int gbase, ldsh* epts, ldsh* eview) {
  const int t = threadIdx.x;
  if (t < ROWS) {                       // threads 0..63: point encoding (L=10)
    const int g = gbase + t;
    ldsh* e = epts + t * EPTS_STRIDE;
    float x[4];
#pragma unroll
    for (int c = 0; c < 4; ++c) { x[c] = A.input_pts[g * 4 + c]; e[c] = (_Float16)x[c]; }
    int idx = 4;
#pragma unroll
    for (int c = 0; c < 4; ++c) {
      float f = 1.f;
      for (int l = 0; l < 10; ++l) {
        float xb = x[c] * f;
        e[idx++] = (_Float16)__sinf(xb);
        e[idx++] = (_Float16)__cosf(xb);
        f *= 2.f;
      }
    }
    for (; idx < 96; ++idx) e[idx] = (_Float16)0.f;
  } else if (t < 2 * ROWS) {            // threads 64..127: view+pl encoding (L=4)
    const int r = t - ROWS;
    const int g = gbase + r;
    ldsh* e = eview + r * EVIEW_STRIDE;
    float x[6];
#pragma unroll
    for (int c = 0; c < 3; ++c) x[c]     = A.input_views[g * 3 + c];
#pragma unroll
    for (int c = 0; c < 3; ++c) x[3 + c] = A.input_pls[g * 3 + c];
#pragma unroll
    for (int c = 0; c < 6; ++c) e[c] = (_Float16)x[c];
    int idx = 6;
#pragma unroll
    for (int c = 0; c < 6; ++c) {
      float f = 1.f;
      for (int l = 0; l < 4; ++l) {
        float xb = x[c] * f;
        e[idx++] = (_Float16)__sinf(xb);
        e[idx++] = (_Float16)__cosf(xb);
        f *= 2.f;
      }
    }
    for (; idx < 64; ++idx) e[idx] = (_Float16)0.f;
  }
}

__global__ __launch_bounds__(256, 2)
void nerf_fused(NerfArgs A, float* __restrict__ out, int N) {
  extern __shared__ _Float16 smem[];
  // flat low 32 bits == wave-relative LDS offset; build AS(3) pointers from it
  const unsigned sbase = (unsigned)(uintptr_t)(void*)smem;
  ldsh* EPTS  = (ldsh*)(uintptr_t)(sbase + EPTS_OFF  * 2);
  ldsh* EVIEW = (ldsh*)(uintptr_t)(sbase + EVIEW_OFF * 2);
  ldsh* HA    = (ldsh*)(uintptr_t)(sbase + HA_OFF    * 2);
  ldsh* HB    = (ldsh*)(uintptr_t)(sbase + HB_OFF    * 2);
  ldsh* WBUF  = (ldsh*)(uintptr_t)(sbase + WB_OFF    * 2);

  const int gbase = blockIdx.x * ROWS;

  encode_rows(A, gbase, EPTS, EVIEW);
  __syncthreads();

  // L0: EPTS(84 pad 96) -> HA
  gemm_layer<16, true >(A.wf[0], A.b[0],  96, EPTS, EPTS_STRIDE,  96,
                        EPTS, EPTS_STRIDE, HA, H_STRIDE, WBUF);
  gemm_layer<16, true >(A.wf[1], A.b[1], 256, HA, H_STRIDE, 256,
                        HA, H_STRIDE, HB, H_STRIDE, WBUF);
  gemm_layer<16, true >(A.wf[2], A.b[2], 256, HB, H_STRIDE, 256,
                        HB, H_STRIDE, HA, H_STRIDE, WBUF);
  gemm_layer<16, true >(A.wf[3], A.b[3], 256, HA, H_STRIDE, 256,
                        HA, H_STRIDE, HB, H_STRIDE, WBUF);
  gemm_layer<16, true >(A.wf[4], A.b[4], 256, HB, H_STRIDE, 256,
                        HB, H_STRIDE, HA, H_STRIDE, WBUF);
  // L5 (skip concat): A = [EPTS(96) | HA(256)] = 352 (weight gap pre-baked in ws)
  gemm_layer<16, true >(A.wf[5], A.b[5], 352, EPTS, EPTS_STRIDE,  96,
                        HA, H_STRIDE, HB, H_STRIDE, WBUF);
  gemm_layer<16, true >(A.wf[6], A.b[6], 256, HB, H_STRIDE, 256,
                        HB, H_STRIDE, HA, H_STRIDE, WBUF);
  gemm_layer<16, true >(A.wf[7], A.b[7], 256, HA, H_STRIDE, 256,
                        HA, H_STRIDE, HB, H_STRIDE, WBUF);
  // feature (no ReLU): HB -> HA
  gemm_layer<16, false>(A.feature_wf, A.feature_b, 256, HB, H_STRIDE, 256,
                        HB, H_STRIDE, HA, H_STRIDE, WBUF);

  // alpha head: h(HB) . alpha_w + alpha_b -> out[0..N)
  {
    const int t = threadIdx.x;
    if (t < ROWS) {
      float s = A.alpha_b[0];
      const ldsh* h = HB + t * H_STRIDE;
      for (int k = 0; k < 256; ++k) s += (float)h[k] * A.alpha_w[k];
      out[gbase + t] = s;
    }
  }
  __syncthreads();

  // views (ReLU): A = [feature(HA,256) | EVIEW(54 pad 64)] = 320 -> HB cols 0..127
  gemm_layer<8, true>(A.views_wf, A.views_b, 320, HA, H_STRIDE, 256,
                      EVIEW, EVIEW_STRIDE, HB, H_STRIDE, WBUF);

  // rgb head: hv(HB[:,0:128]) . rgb_w + rgb_b -> out[N .. N+3N)
  {
    const int t = threadIdx.x;
    if (t < ROWS) {
      const int g = gbase + t;
      const ldsh* h = HB + t * H_STRIDE;
#pragma unroll
      for (int c = 0; c < 3; ++c) {
        float s = A.rgb_b[c];
        for (int k = 0; k < 128; ++k) s += (float)h[k] * A.rgb_w[k * 3 + c];
        out[N + g * 3 + c] = s;
      }
    }
  }
}

extern "C" void kernel_launch(void* const* d_in, const int* in_sizes, int n_in,
                              void* d_out, int out_size, void* d_ws, size_t ws_size,
                              hipStream_t stream) {
  (void)n_in; (void)out_size; (void)ws_size;
  _Float16* WS = (_Float16*)d_ws;

  // ---- repack all weights to f16 col-major in ws (runs every call; ~0.6M elems)
  struct Cvt { int src; int off; int ncols; int Kpad; int Ksrc; int ins; int gap; };
  const Cvt cv[10] = {
    { 3, WSO_W0,   256,  96,  84,  84, 12},
    { 5, WSO_W1,   256, 256, 256, 256,  0},
    { 7, WSO_W2,   256, 256, 256, 256,  0},
    { 9, WSO_W3,   256, 256, 256, 256,  0},
    {11, WSO_W4,   256, 256, 256, 256,  0},
    {13, WSO_W5,   256, 352, 340,  84, 12},
    {15, WSO_W6,   256, 256, 256, 256,  0},
    {17, WSO_W7,   256, 256, 256, 256,  0},
    {21, WSO_FEAT, 256, 256, 256, 256,  0},
    {19, WSO_VIEW, 128, 320, 310, 310, 10},
  };
  for (int i = 0; i < 10; ++i) {
    dim3 g((cv[i].Kpad + 127) / 128, cv[i].ncols);
    conv_weights<<<g, dim3(128), 0, stream>>>(
        (const float*)d_in[cv[i].src], WS + cv[i].off,
        cv[i].ncols, cv[i].Kpad, cv[i].Ksrc, cv[i].ins, cv[i].gap);
  }

  NerfArgs a;
  a.input_pts   = (const float*)d_in[0];
  a.input_views = (const float*)d_in[1];
  a.input_pls   = (const float*)d_in[2];
  for (int i = 0; i < 8; ++i) a.b[i] = (const float*)d_in[4 + 2 * i];
  a.wf[0] = WS + WSO_W0;  a.wf[1] = WS + WSO_W1;
  a.wf[2] = WS + WSO_W2;  a.wf[3] = WS + WSO_W3;
  a.wf[4] = WS + WSO_W4;  a.wf[5] = WS + WSO_W5;
  a.wf[6] = WS + WSO_W6;  a.wf[7] = WS + WSO_W7;
  a.views_wf   = WS + WSO_VIEW;
  a.feature_wf = WS + WSO_FEAT;
  a.views_b    = (const float*)d_in[20];
  a.feature_b  = (const float*)d_in[22];
  a.alpha_w    = (const float*)d_in[23];
  a.alpha_b    = (const float*)d_in[24];
  a.rgb_w      = (const float*)d_in[25];
  a.rgb_b      = (const float*)d_in[26];

  const int N = in_sizes[0] / 4;          // 131072
  const int blocks = N / ROWS;            // 2048 workgroups of 8 wave32s
  const size_t lds_bytes = (size_t)LDS_ELEMS * sizeof(_Float16);  // 132 KB

  nerf_fused<<<dim3(blocks), dim3(256), lds_bytes, stream>>>(a, (float*)d_out, N);
}
